// PointNetMSG_42236708389458
// MI455X (gfx1250) — matrix operand order repaired and checked by reading, not compiled, not run on hardware
//
#include <hip/hip_runtime.h>
#include <hip/hip_bf16.h>

typedef __attribute__((ext_vector_type(16))) _Float16 v16h;
typedef __attribute__((ext_vector_type(8)))  _Float16 v8h;
typedef __attribute__((ext_vector_type(8)))  float    v8f;
typedef __attribute__((ext_vector_type(4)))  unsigned int v4u;
typedef __attribute__((ext_vector_type(8)))  unsigned int v8u;

#define BATCH  4
#define NPTS   16384
#define NCENT  1024
#define INCH   32
#define FEATCH 320
#define BNEPS  1e-5f

// ---------------------------------------------------------------------------
// Tensor Data Mover: 1-D contiguous copy global -> LDS (ISA ch.8 D# layout).
// lds_off = byte offset within the workgroup's dynamic LDS; nbytes mult of 8.
// Issued by one wave; completion via s_wait_tensorcnt + workgroup barrier.
// ---------------------------------------------------------------------------
__device__ __attribute__((always_inline)) inline void tdm_load_1d(
    unsigned lds_off, const void* gptr, unsigned nbytes) {
  const unsigned long long ga = (unsigned long long)(uintptr_t)gptr;
  const unsigned n8 = nbytes >> 3;  // 8-byte elements (data_size=3)
  v4u g0;
  g0.x = 1u;                                        // count=1, user mode
  g0.y = lds_off;                                   // lds_addr
  g0.z = (unsigned)(ga & 0xffffffffu);              // global_addr[31:0]
  g0.w = (unsigned)((ga >> 32) & 0x01ffffffu)       // global_addr[56:32]
         | (2u << 30);                              // type=2 (image)
  v8u g1;
  g1.s0 = 3u << 16;                 // workgroup_mask=0, data_size=3 (8B)
  g1.s1 = (n8 & 0xffffu) << 16;     // tensor_dim0[15:0]
  g1.s2 = ((n8 >> 16) & 0xffffu)    // tensor_dim0[31:16]
          | (1u << 16);             // tensor_dim1 = 1
  g1.s3 = (n8 & 0xffffu) << 16;     // tensor_dim1 hi=0 | tile_dim0 = n8
  g1.s4 = 1u;                       // tile_dim1=1, tile_dim2=0
  g1.s5 = n8;                       // tensor_dim0_stride[31:0]
  g1.s6 = 0u;                       // stride0 hi | stride1 lo
  g1.s7 = 0u;
  asm volatile("tensor_load_to_lds %0, %1" :: "s"(g0), "s"(g1) : "memory");
}

// ---------------------------------------------------------------------------
// Prep: f32 weights [cout][cin] -> zero-padded f16 [cout][cinp]
// ---------------------------------------------------------------------------
__global__ void prep_weights(const float* __restrict__ W, int cout, int cin,
                             int cinp, _Float16* __restrict__ Wh) {
  int i = blockIdx.x * blockDim.x + threadIdx.x;
  if (i >= cout * cinp) return;
  int r = i / cinp, c = i % cinp;
  float v = (c < cin) ? W[r * cin + c] : 0.0f;
  Wh[i] = (_Float16)v;
}

// Fold BatchNorm (eval) + conv bias: y = dot*alpha + delta
__global__ void prep_bn(const float* __restrict__ b, const float* __restrict__ g,
                        const float* __restrict__ be, const float* __restrict__ me,
                        const float* __restrict__ va, int cout,
                        float* __restrict__ ad) {
  int i = blockIdx.x * blockDim.x + threadIdx.x;
  if (i >= cout) return;
  float s = g[i] * rsqrtf(va[i] + BNEPS);
  ad[i]        = s;
  ad[cout + i] = (b[i] - me[i]) * s + be[i];
}

// ---------------------------------------------------------------------------
// FPS: one block per batch, 1024 threads, 16 register-resident points/thread,
// LDS tree argmax (ties -> lowest index, matching jnp.argmax).
// ---------------------------------------------------------------------------
__global__ __launch_bounds__(1024) void fps_kernel(const float* __restrict__ xyz,
                                                   float* __restrict__ new_xyz) {
  const int b = blockIdx.x;
  const int tid = threadIdx.x;
  constexpr int PT = NPTS / 1024;  // 16
  const float* base = xyz + (size_t)b * NPTS * 3;
  float px[PT], py[PT], pz[PT], dist[PT];
#pragma unroll
  for (int j = 0; j < PT; ++j) {
    int i = j * 1024 + tid;
    px[j] = base[i * 3 + 0];
    py[j] = base[i * 3 + 1];
    pz[j] = base[i * 3 + 2];
    dist[j] = 1e10f;
  }
  __shared__ float s_val[1024];
  __shared__ int   s_idx[1024];
  __shared__ float cc[3];
  int cur = 0;
  for (int s = 0; s < NCENT; ++s) {
    if (tid == 0) {
      float x0 = base[cur * 3 + 0], y0 = base[cur * 3 + 1], z0 = base[cur * 3 + 2];
      size_t o = ((size_t)b * NCENT + s) * 3;
      new_xyz[o + 0] = x0; new_xyz[o + 1] = y0; new_xyz[o + 2] = z0;
      cc[0] = x0; cc[1] = y0; cc[2] = z0;
    }
    __syncthreads();
    float cx = cc[0], cy = cc[1], cz = cc[2];
    float bv = -1.0f; int bi = tid;
#pragma unroll
    for (int j = 0; j < PT; ++j) {
      float dx = px[j] - cx, dy = py[j] - cy, dz = pz[j] - cz;
      float d = dx * dx + dy * dy + dz * dz;
      if (d < dist[j]) dist[j] = d;
      if (dist[j] > bv) { bv = dist[j]; bi = j * 1024 + tid; }
    }
    s_val[tid] = bv; s_idx[tid] = bi;
    for (int off = 512; off > 0; off >>= 1) {
      __syncthreads();
      if (tid < off) {
        float v2 = s_val[tid + off]; int i2 = s_idx[tid + off];
        if (v2 > s_val[tid] || (v2 == s_val[tid] && i2 < s_idx[tid])) {
          s_val[tid] = v2; s_idx[tid] = i2;
        }
      }
    }
    __syncthreads();
    cur = s_idx[0];
  }
}

// ---------------------------------------------------------------------------
// A-stationary WMMA MLP layer. A in LDS (stride STRIDE halves); each wave owns
// M-tiles, keeps its A fragments (16x32, contiguous k = 16*hi + s per lane) in VGPRs,
// sweeps N-tiles streaming B fragments from the TDM-staged LDS weight blob.
// Epilogue: folded BN (alpha/delta) + PReLU, stored f16 to the ping-pong buf.
// ---------------------------------------------------------------------------
template <int ROWS, int CIN, int COUT, int STRIDE>
__device__ __attribute__((always_inline)) inline void layer_wmma(
    const _Float16* Asrc, _Float16* Adst, const _Float16* Wl,
    const float* __restrict__ ad, float slope, int wave, int lane) {
  constexpr int MT = ROWS / 16;
  constexpr int NT = COUT / 16;
  constexpr int KS = CIN / 32;
  const int half16 = lane & 15;
  const int hi = lane >> 4;  // 0 or 1
  for (int mt = wave; mt < MT; mt += 8) {
    const int m0 = mt * 16;
    v16h a[KS];
    const _Float16* arow = Asrc + (m0 + half16) * STRIDE + hi * 16;
#pragma unroll
    for (int ks = 0; ks < KS; ++ks) {
      v8h alo = *(const v8h*)(arow + ks * 32);       // K = kk + hi*16 + 0..7
      v8h ahi = *(const v8h*)(arow + ks * 32 + 8);   // K = kk + hi*16 + 8..15
#pragma unroll
      for (int i = 0; i < 8; ++i) { a[ks][i] = alo[i]; a[ks][8 + i] = ahi[i]; }
    }
    for (int nt = 0; nt < NT; ++nt) {
      const int n0 = nt * 16;
      const _Float16* wrow = Wl + (n0 + half16) * CIN + hi * 16;
      v8f c = {};
#pragma unroll
      for (int ks = 0; ks < KS; ++ks) {
        v16h bm = *(const v16h*)(wrow + ks * 32);    // K = kk + hi*16 + 0..15
        c = __builtin_amdgcn_wmma_f32_16x16x32_f16(false, a[ks], false, bm,
                                                   (short)0, c, false, false);
      }
      const int n = n0 + half16;
      const float al = ad[n];
      const float de = ad[COUT + n];
#pragma unroll
      for (int r = 0; r < 8; ++r) {
        float v = c[r] * al + de;
        v = v >= 0.0f ? v : slope * v;
        Adst[(m0 + r + hi * 8) * STRIDE + n] = (_Float16)v;
      }
    }
  }
}

// ---------------------------------------------------------------------------
// Fused per-scale kernel: TDM weight prefetch || ball query -> gather ->
// 3 WMMA layers (LDS ping-pong, LDS weights) -> max-pool -> f16 concat buffer.
// ROWS = TC*K = 256; 256 threads / 8 waves.
// ---------------------------------------------------------------------------
template <int K, int TC, int C0, int C1, int C2, int STRIDE>
__global__ __launch_bounds__(256) void scale_kernel(
    const float* __restrict__ xyz, const float* __restrict__ feat,
    const float* __restrict__ new_xyz,
    const _Float16* __restrict__ Wg,   // packed f16 weights for all 3 layers
    const float* __restrict__ ad0, const float* __restrict__ ad1,
    const float* __restrict__ ad2,
    const float* __restrict__ a0p, const float* __restrict__ a1p,
    const float* __restrict__ a2p,
    float r2, _Float16* __restrict__ featsH, int scale_off) {
  constexpr int ROWS = TC * K;  // 256
  constexpr unsigned WOFF = 2u * ROWS * STRIDE * 2u + ROWS * 4u;  // bytes
  constexpr unsigned WHALVES = C0 * 64 + C1 * C0 + C2 * C1;
  extern __shared__ char smem[];
  _Float16* buf0 = (_Float16*)smem;
  _Float16* buf1 = buf0 + ROWS * STRIDE;
  int* nbr = (int*)(buf1 + ROWS * STRIDE);
  _Float16* wlds = (_Float16*)(smem + WOFF);
  const int tid = threadIdx.x;
  const int wave = tid >> 5, lane = tid & 31;
  const int g0 = blockIdx.x * TC;

  // ---- Kick the TDM: weights DMA into LDS overlaps ball query + gather.
  if (wave == 0) tdm_load_1d(WOFF, Wg, WHALVES * 2u);

  // ---- Phase 1: ordered ball query (first K in-radius indices, ascending),
  //               one wave per center, wave32 ballot + prefix popcount.
  for (int c = wave; c < TC; c += 8) {
    const int g = g0 + c;
    const int b = g / NCENT;
    const float cx = new_xyz[g * 3 + 0];
    const float cy = new_xyz[g * 3 + 1];
    const float cz = new_xyz[g * 3 + 2];
    const float* pb = xyz + (size_t)b * NPTS * 3;
    int cnt = 0;
    int firstIdx = NPTS - 1;  // pad target when zero hits (matches jnp clamp)
    const unsigned lt = (1u << lane) - 1u;
    for (int bs = 0; bs < NPTS && cnt < K; bs += 32) {
      const int i = bs + lane;
      const float dx = pb[i * 3 + 0] - cx;
      const float dy = pb[i * 3 + 1] - cy;
      const float dz = pb[i * 3 + 2] - cz;
      const bool hit = (dx * dx + dy * dy + dz * dz) <= r2;
      const unsigned m = __builtin_amdgcn_ballot_w32(hit);
      if (cnt == 0 && m) firstIdx = bs + __builtin_ctz(m);
      const int pos = cnt + __popc(m & lt);
      if (hit && pos < K) nbr[c * K + pos] = i;
      cnt += __popc(m);
    }
    const int nv = cnt < K ? cnt : K;
    for (int j = nv + lane; j < K; j += 32) nbr[c * K + j] = firstIdx;
  }
  __syncthreads();

  // ---- Phase 2: gather rows -> buf0 f16: [dx,dy,dz, feat(32), zeros..63]
  for (int r = tid; r < ROWS; r += 256) {
    const int c = r / K;
    const int g = g0 + c;
    const int b = g / NCENT;
    const int i = nbr[r];
    const float* pb = xyz + ((size_t)b * NPTS + i) * 3;
    _Float16* row = buf0 + r * STRIDE;
    row[0] = (_Float16)(pb[0] - new_xyz[g * 3 + 0]);
    row[1] = (_Float16)(pb[1] - new_xyz[g * 3 + 1]);
    row[2] = (_Float16)(pb[2] - new_xyz[g * 3 + 2]);
    const float* fb = feat + ((size_t)b * NPTS + i) * INCH;
#pragma unroll
    for (int ch = 0; ch < INCH; ++ch) row[3 + ch] = (_Float16)fb[ch];
#pragma unroll
    for (int ch = 3 + INCH; ch < 64; ++ch) row[ch] = (_Float16)0.0f;
  }
  // Weights must have landed before layer 1.
  if (wave == 0) __builtin_amdgcn_s_wait_tensorcnt(0);
  __syncthreads();

  // ---- Phase 3: 3 WMMA layers, LDS ping-pong, LDS-resident weights
  const _Float16* wl0 = wlds;
  const _Float16* wl1 = wl0 + C0 * 64;
  const _Float16* wl2 = wl1 + C1 * C0;
  layer_wmma<ROWS, 64, C0, STRIDE>(buf0, buf1, wl0, ad0, a0p[0], wave, lane);
  __syncthreads();
  layer_wmma<ROWS, C0, C1, STRIDE>(buf1, buf0, wl1, ad1, a1p[0], wave, lane);
  __syncthreads();
  layer_wmma<ROWS, C1, C2, STRIDE>(buf0, buf1, wl2, ad2, a2p[0], wave, lane);
  __syncthreads();

  // ---- Phase 4: max-pool over K samples -> f16 concat feature buffer
  for (int p = tid; p < TC * C2; p += 256) {
    const int c = p / C2, ch = p % C2;
    const _Float16* rb = buf1 + (c * K) * STRIDE + ch;
    float mx = -3.0e38f;
    for (int j = 0; j < K; ++j) {
      const float v = (float)rb[j * STRIDE];
      mx = v > mx ? v : mx;
    }
    featsH[(size_t)(g0 + c) * FEATCH + scale_off + ch] = (_Float16)mx;
  }
}

// ---------------------------------------------------------------------------
// Final MLP: [4096 x 320] -> 256 -> 256. Layer1: A-stationary, A+W0 streamed
// from global while TDM prefetches W1 (128 KB) into LDS. Layer2: A from LDS
// ping buffer, B from LDS. 256 KB dynamic LDS (CDNA5 320 KB/WGP).
// ---------------------------------------------------------------------------
__global__ __launch_bounds__(256) void final_kernel(
    const _Float16* __restrict__ A,  // [4096][320]
    const _Float16* __restrict__ W0, const float* __restrict__ ad0,
    const float* __restrict__ a0p,
    const _Float16* __restrict__ W1, const float* __restrict__ ad1,
    const float* __restrict__ a1p,
    float* __restrict__ out) {
  constexpr unsigned W1OFF = 256u * 256u * 2u;  // after activation buffer
  extern __shared__ char smem[];
  _Float16* buf = (_Float16*)smem;              // 256 rows x 256 ch f16
  _Float16* w1lds = (_Float16*)(smem + W1OFF);  // 256 x 256 f16 weights
  const int tid = threadIdx.x, wave = tid >> 5, lane = tid & 31;
  const int half16 = lane & 15, hi = lane >> 4;
  const int m0base = blockIdx.x * 256;
  const float s0 = a0p[0], s1 = a1p[0];

  if (wave == 0) tdm_load_1d(W1OFF, W1, 256u * 256u * 2u);

  // Layer 1: K=320, A-stationary, W0 streamed from global (L2-resident)
  for (int mt = wave; mt < 16; mt += 8) {
    const int m0 = mt * 16;
    v16h a[10];
    const _Float16* arow = A + (size_t)(m0base + m0 + half16) * FEATCH + hi * 16;
#pragma unroll
    for (int ks = 0; ks < 10; ++ks) {
      v8h alo = *(const v8h*)(arow + ks * 32);
      v8h ahi = *(const v8h*)(arow + ks * 32 + 8);
#pragma unroll
      for (int i = 0; i < 8; ++i) { a[ks][i] = alo[i]; a[ks][8 + i] = ahi[i]; }
    }
    for (int nt = 0; nt < 16; ++nt) {
      const int n0 = nt * 16;
      const _Float16* wrow = W0 + (size_t)(n0 + half16) * FEATCH + hi * 16;
      v8f c = {};
#pragma unroll
      for (int ks = 0; ks < 10; ++ks) {
        v16h bm = *(const v16h*)(wrow + ks * 32);
        c = __builtin_amdgcn_wmma_f32_16x16x32_f16(false, a[ks], false, bm,
                                                   (short)0, c, false, false);
      }
      const int n = n0 + half16;
      const float al = ad0[n], de = ad0[256 + n];
#pragma unroll
      for (int r = 0; r < 8; ++r) {
        float v = c[r] * al + de;
        v = v >= 0.0f ? v : s0 * v;
        buf[(m0 + r + hi * 8) * 256 + n] = (_Float16)v;
      }
    }
  }
  if (wave == 0) __builtin_amdgcn_s_wait_tensorcnt(0);
  __syncthreads();

  // Layer 2: K=256, A from LDS, B from TDM-staged LDS, f32 output
  for (int mt = wave; mt < 16; mt += 8) {
    const int m0 = mt * 16;
    v16h a[8];
    const _Float16* arow = buf + (m0 + half16) * 256 + hi * 16;
#pragma unroll
    for (int ks = 0; ks < 8; ++ks) {
      v8h alo = *(const v8h*)(arow + ks * 32);
      v8h ahi = *(const v8h*)(arow + ks * 32 + 8);
#pragma unroll
      for (int i = 0; i < 8; ++i) { a[ks][i] = alo[i]; a[ks][8 + i] = ahi[i]; }
    }
    for (int nt = 0; nt < 16; ++nt) {
      const int n0 = nt * 16;
      const _Float16* wrow = w1lds + (n0 + half16) * 256 + hi * 16;
      v8f c = {};
#pragma unroll
      for (int ks = 0; ks < 8; ++ks) {
        v16h bm = *(const v16h*)(wrow + ks * 32);
        c = __builtin_amdgcn_wmma_f32_16x16x32_f16(false, a[ks], false, bm,
                                                   (short)0, c, false, false);
      }
      const int n = n0 + half16;
      const float al = ad1[n], de = ad1[256 + n];
#pragma unroll
      for (int r = 0; r < 8; ++r) {
        float v = c[r] * al + de;
        v = v >= 0.0f ? v : s1 * v;
        out[(size_t)(m0base + m0 + r + hi * 8) * 256 + n] = v;
      }
    }
  }
}

// ---------------------------------------------------------------------------
// Host launch
// ---------------------------------------------------------------------------
extern "C" void kernel_launch(void* const* d_in, const int* in_sizes, int n_in,
                              void* d_out, int out_size, void* d_ws,
                              size_t ws_size, hipStream_t stream) {
  (void)out_size; (void)ws_size;

  // Resolve input binding robustly from sizes (dict insertion order vs sorted).
  const float* xyz;
  const float* feat;
  int pstart;
  if (in_sizes[0] == BATCH * NPTS * 3) {        // xyz, features, params...
    xyz = (const float*)d_in[0];
    feat = (const float*)d_in[1];
    pstart = 2;
  } else {                                       // features, params..., xyz
    feat = (const float*)d_in[0];
    xyz = (const float*)d_in[n_in - 1];
    pstart = 1;
  }
  const bool final_first = (in_sizes[pstart] == 256 * FEATCH);

  // Canonical layer list: s1(3), s2(3), s3(3), final(2)
  static const int cins[11]  = {35, 32, 32, 35, 64, 64, 35, 64, 96, FEATCH, 256};
  static const int couts[11] = {32, 32, 64, 64, 64, 128, 64, 96, 128, 256, 256};
  static const int order_ff[11] = {9, 10, 0, 1, 2, 3, 4, 5, 6, 7, 8};
  const float* P[11][7];  // W, b, gamma, beta, mean, var, a
  int idx = pstart;
  for (int s = 0; s < 11; ++s) {
    const int L = final_first ? order_ff[s] : s;
    for (int j = 0; j < 7; ++j) P[L][j] = (const float*)d_in[idx++];
  }

  // Workspace carve. All Wh blocks are 256-byte multiples, carved back-to-back
  // so each scale's 3 layers form one contiguous blob for a single TDM copy.
  char* ws = (char*)d_ws;
  size_t off = 0;
  auto carve = [&](size_t bytes) -> char* {
    char* p = ws + off;
    off = (off + bytes + 255) & ~(size_t)255;
    return p;
  };
  _Float16* featsH = (_Float16*)carve((size_t)BATCH * NCENT * FEATCH * 2);
  _Float16* Wh[11];
  float* AD[11];
  int cinp[11];
  for (int L = 0; L < 11; ++L) {
    cinp[L] = ((cins[L] + 31) / 32) * 32;
    Wh[L] = (_Float16*)carve((size_t)couts[L] * cinp[L] * 2);
  }
  for (int L = 0; L < 11; ++L) AD[L] = (float*)carve((size_t)couts[L] * 2 * 4);

  float* out_xyz = (float*)d_out;
  float* out_x = out_xyz + (size_t)BATCH * NCENT * 3;

  // 1) Weight/BN prep
  for (int L = 0; L < 11; ++L) {
    const int tot = couts[L] * cinp[L];
    prep_weights<<<(tot + 255) / 256, 256, 0, stream>>>(P[L][0], couts[L],
                                                        cins[L], cinp[L], Wh[L]);
    prep_bn<<<(couts[L] + 63) / 64, 64, 0, stream>>>(P[L][1], P[L][2], P[L][3],
                                                     P[L][4], P[L][5], couts[L],
                                                     AD[L]);
  }

  // 2) FPS -> new_xyz (d_out)
  fps_kernel<<<BATCH, 1024, 0, stream>>>(xyz, out_xyz);

  // 3) Fused scale kernels (dynamic LDS > 64 KB leverages 320 KB/WGP)
  {
    const size_t sh = (size_t)2 * 256 * 64 * 2 + 256 * 4 + 10240;
    (void)hipFuncSetAttribute(
        reinterpret_cast<const void*>(scale_kernel<16, 16, 32, 32, 64, 64>),
        hipFuncAttributeMaxDynamicSharedMemorySize, (int)sh);
    scale_kernel<16, 16, 32, 32, 64, 64>
        <<<BATCH * NCENT / 16, 256, sh, stream>>>(
            xyz, feat, out_xyz, Wh[0], AD[0], AD[1], AD[2],
            P[0][6], P[1][6], P[2][6], 0.1f * 0.1f, featsH, 0);
  }
  {
    const size_t sh = (size_t)2 * 256 * 64 * 2 + 256 * 4 + 32768;
    (void)hipFuncSetAttribute(
        reinterpret_cast<const void*>(scale_kernel<32, 8, 64, 64, 128, 64>),
        hipFuncAttributeMaxDynamicSharedMemorySize, (int)sh);
    scale_kernel<32, 8, 64, 64, 128, 64>
        <<<BATCH * NCENT / 8, 256, sh, stream>>>(
            xyz, feat, out_xyz, Wh[3], AD[3], AD[4], AD[5],
            P[3][6], P[4][6], P[5][6], 0.2f * 0.2f, featsH, 64);
  }
  {
    const size_t sh = (size_t)2 * 256 * 96 * 2 + 256 * 4 + 45056;
    (void)hipFuncSetAttribute(
        reinterpret_cast<const void*>(scale_kernel<64, 4, 64, 96, 128, 96>),
        hipFuncAttributeMaxDynamicSharedMemorySize, (int)sh);
    scale_kernel<64, 4, 64, 96, 128, 96>
        <<<BATCH * NCENT / 4, 256, sh, stream>>>(
            xyz, feat, out_xyz, Wh[6], AD[6], AD[7], AD[8],
            P[6][6], P[7][6], P[8][6], 0.4f * 0.4f, featsH, 192);
  }

  // 4) Final MLP -> d_out x-region
  {
    const size_t sh = (size_t)256 * 256 * 2 * 2;  // act buf + TDM'd W1
    (void)hipFuncSetAttribute(reinterpret_cast<const void*>(final_kernel),
                              hipFuncAttributeMaxDynamicSharedMemorySize,
                              (int)sh);
    final_kernel<<<BATCH * NCENT / 256, 256, sh, stream>>>(
        featsH, Wh[9], AD[9], P[9][6], Wh[10], AD[10], P[10][6], out_x);
  }
}
